// SpikingUNetRNN_24773371363548
// MI455X (gfx1250) — matrix operand order repaired and checked by reading, not compiled, not run on hardware
//
#include <hip/hip_runtime.h>
#include <math.h>

typedef __attribute__((ext_vector_type(16))) __bf16 v16bf;
typedef __attribute__((ext_vector_type(8)))  __bf16 v8bf;
typedef __attribute__((ext_vector_type(8)))  float  v8f;

static constexpr int T = 16;
static constexpr int BATCH = 32;

// ---------------------------------------------------------------- utilities
__global__ void zero_u32_kernel(unsigned int* p, int n) {
  int i = blockIdx.x * blockDim.x + threadIdx.x;
  if (i < n) p[i] = 0u;
}

__global__ void cvt_bf16_kernel(const float* __restrict__ s, __bf16* __restrict__ d, int n) {
  int i = blockIdx.x * blockDim.x + threadIdx.x;
  if (i < n) d[i] = (__bf16)s[i];
}

// ------------------------------------------------- conv 3x3 stride2 pad1
__global__ void conv3x3_s2_kernel(const float* __restrict__ in, const float* __restrict__ w,
                                  float* __restrict__ out,
                                  int N, int Cin, int Hin, int Win, int Cout) {
  const int Ho = Hin >> 1, Wo = Win >> 1;
  int idx = blockIdx.x * blockDim.x + threadIdx.x;
  int total = N * Cout * Ho * Wo;
  if (idx >= total) return;
  int ox = idx % Wo;
  int oy = (idx / Wo) % Ho;
  int co = (idx / (Wo * Ho)) % Cout;
  int n  = idx / (Wo * Ho * Cout);
  float acc = 0.f;
  for (int ci = 0; ci < Cin; ++ci)
    for (int ky = 0; ky < 3; ++ky) {
      int iy = 2 * oy - 1 + ky;
      if (iy < 0 || iy >= Hin) continue;
      for (int kx = 0; kx < 3; ++kx) {
        int ix = 2 * ox - 1 + kx;
        if (ix < 0 || ix >= Win) continue;
        acc += in[((n * Cin + ci) * Hin + iy) * Win + ix] * w[((co * Cin + ci) * 3 + ky) * 3 + kx];
      }
    }
  out[idx] = acc;
}

// ------------------------------------------- per-channel sum / sumsq (BN train stats)
__global__ void chan_stats_kernel(const float* __restrict__ x, int N, int C, int HW,
                                  float* __restrict__ sum, float* __restrict__ sumsq) {
  const int c = blockIdx.y;
  const int tid = threadIdx.x;
  const int total = N * HW;
  float s = 0.f, q = 0.f;
  for (int i = blockIdx.x * blockDim.x + tid; i < total; i += gridDim.x * blockDim.x) {
    int n = i / HW, p = i - n * HW;
    float v = x[(n * C + c) * HW + p];
    s += v; q += v * v;
  }
  __shared__ float sh_s[256], sh_q[256];
  sh_s[tid] = s; sh_q[tid] = q;
  __syncthreads();
  for (int w = 128; w > 0; w >>= 1) {
    if (tid < w) { sh_s[tid] += sh_s[tid + w]; sh_q[tid] += sh_q[tid + w]; }
    __syncthreads();
  }
  if (tid == 0) { atomicAdd(&sum[c], sh_s[0]); atomicAdd(&sumsq[c], sh_q[0]); }
}

// --------------------------------- fused BatchNorm + LIF scan over T (hard reset)
__global__ void bn_lif_kernel(const float* __restrict__ xin,
                              const float* __restrict__ sum, const float* __restrict__ sumsq,
                              const float* __restrict__ g, const float* __restrict__ beta,
                              int C, int HW, float count,
                              float* __restrict__ out_f32, __bf16* __restrict__ out_bf16) {
  int idx = blockIdx.x * blockDim.x + threadIdx.x;
  int total = BATCH * C * HW;
  if (idx >= total) return;
  int p = idx % HW;
  int c = (idx / HW) % C;
  int b = idx / (HW * C);
  float mean = sum[c] / count;
  float var  = sumsq[c] / count - mean * mean;
  float sc = g[c] * rsqrtf(var + 1e-5f);
  float sh = beta[c] - mean * sc;
  float v = 0.f;
  const int chw = C * HW;
  for (int t = 0; t < T; ++t) {
    int off = (t * BATCH + b) * chw + c * HW + p;
    float xv = xin[off] * sc + sh;
    v = 0.5f * (v + xv);                       // charge, tau=2
    float s = (v >= 1.0f) ? 1.0f : 0.0f;       // fire
    v = (v >= 1.0f) ? 0.0f : v;                // hard reset
    if (out_f32) out_f32[off] = s;
    else         out_bf16[off] = (__bf16)s;
  }
}

// ------------------------------------------------- WMMA helpers (bf16 16x16x32)
__device__ inline v16bf cat8(v8bf lo, v8bf hi) {
  return __builtin_shufflevector(lo, hi, 0,1,2,3,4,5,6,7,8,9,10,11,12,13,14,15);
}

// Recurrent step: z[32,4096] = cat(enc[t], y[t-1])[32,8192] @ rc_w^T + b ; fused LIF.
// One wave per 16-col tile; 2 WMMAs (rows 0..15 / 16..31) share B fragment.
__global__ __launch_bounds__(32)
void gemm_rc_lif_kernel(const __bf16* __restrict__ encA,    // [32,4096] spikes at t
                        const __bf16* __restrict__ yprev,   // [32,4096] spikes at t-1
                        const __bf16* __restrict__ Wb,      // [4096,8192] bf16
                        const float*  __restrict__ bias,    // [4096]
                        float*  __restrict__ vstate,        // [32,4096]
                        __bf16* __restrict__ lat_t) {       // [32,4096] out spikes
  const int lane = threadIdx.x;
  const int n0 = blockIdx.x * 16;
  const int mr   = lane & 15;          // A row / B column within tile
  const int ksubA = (lane >> 4) * 16;  // A K half select
  const int koffB = (lane >> 4) * 16;  // B K half select
  const __bf16* brow = Wb + (n0 + mr) * 8192 + koffB;
  v8f acc0 = {}; v8f acc1 = {};
  const __bf16* srcs[2] = { encA, yprev };
  for (int h = 0; h < 2; ++h) {
    const __bf16* src = srcs[h];
    const __bf16* bh = brow + h * 4096;
    for (int k0 = 0; k0 < 4096; k0 += 32) {
      __builtin_prefetch(bh + k0 + 512, 0, 1);  // stream next weight chunk
      const __bf16* a0p = src + mr * 4096 + k0 + ksubA;
      const __bf16* a1p = a0p + 16 * 4096;
      v16bf a0 = cat8(*(const v8bf*)a0p, *(const v8bf*)(a0p + 8));
      v16bf a1 = cat8(*(const v8bf*)a1p, *(const v8bf*)(a1p + 8));
      v16bf bb = *(const v16bf*)(bh + k0);
      acc0 = __builtin_amdgcn_wmma_f32_16x16x32_bf16(false, a0, false, bb, (short)0, acc0, false, false);
      acc1 = __builtin_amdgcn_wmma_f32_16x16x32_bf16(false, a1, false, bb, (short)0, acc1, false, false);
    }
  }
  const int n = n0 + mr;
  const int mbase = (lane >> 4) * 8;
  const float bn = bias[n];
  #pragma unroll
  for (int r = 0; r < 8; ++r) {
    int m = mbase + r;
    {
      float z = acc0[r] + bn;
      float v = vstate[m * 4096 + n];
      v = 0.5f * (v + z);
      float s = (v >= 1.0f) ? 1.0f : 0.0f;
      vstate[m * 4096 + n] = (v >= 1.0f) ? 0.0f : v;
      lat_t[m * 4096 + n] = (__bf16)s;
    }
    {
      int m2 = m + 16;
      float z = acc1[r] + bn;
      float v = vstate[m2 * 4096 + n];
      v = 0.5f * (v + z);
      float s = (v >= 1.0f) ? 1.0f : 0.0f;
      vstate[m2 * 4096 + n] = (v >= 1.0f) ? 0.0f : v;
      lat_t[m2 * 4096 + n] = (__bf16)s;
    }
  }
}

// Decoder GEMM: out[512,4096] = lat[512,4096] @ dec_w^T   (dec_w [4096,4096] bf16)
__global__ __launch_bounds__(32)
void gemm_dec_kernel(const __bf16* __restrict__ A, const __bf16* __restrict__ Wb,
                     float* __restrict__ out) {
  const int lane = threadIdx.x;
  const int n0 = blockIdx.x * 16;
  const int m0 = blockIdx.y * 32;
  const int mr = lane & 15;
  const int ksubA = (lane >> 4) * 16;
  const int koffB = (lane >> 4) * 16;
  const __bf16* arow0 = A + (m0 + mr) * 4096;
  const __bf16* arow1 = arow0 + 16 * 4096;
  const __bf16* brow  = Wb + (n0 + mr) * 4096 + koffB;
  v8f acc0 = {}; v8f acc1 = {};
  for (int k0 = 0; k0 < 4096; k0 += 32) {
    v16bf a0 = cat8(*(const v8bf*)(arow0 + k0 + ksubA), *(const v8bf*)(arow0 + k0 + ksubA + 8));
    v16bf a1 = cat8(*(const v8bf*)(arow1 + k0 + ksubA), *(const v8bf*)(arow1 + k0 + ksubA + 8));
    v16bf bb = *(const v16bf*)(brow + k0);
    acc0 = __builtin_amdgcn_wmma_f32_16x16x32_bf16(false, a0, false, bb, (short)0, acc0, false, false);
    acc1 = __builtin_amdgcn_wmma_f32_16x16x32_bf16(false, a1, false, bb, (short)0, acc1, false, false);
  }
  const int n = n0 + mr;
  const int mbase = (lane >> 4) * 8;
  #pragma unroll
  for (int r = 0; r < 8; ++r) {
    out[(m0 + mbase + r) * 4096 + n]      = acc0[r];
    out[(m0 + 16 + mbase + r) * 4096 + n] = acc1[r];
  }
}

// LIF scan over decoder pre-activations [T,B,4096] -> f32 spikes
__global__ void lif_dec_kernel(const float* __restrict__ z, float* __restrict__ spk) {
  int idx = blockIdx.x * blockDim.x + threadIdx.x;
  if (idx >= BATCH * 4096) return;
  int b = idx >> 12, f = idx & 4095;
  float v = 0.f;
  for (int t = 0; t < T; ++t) {
    int off = (t * BATCH + b) * 4096 + f;
    v = 0.5f * (v + z[off]);
    float s = (v >= 1.0f) ? 1.0f : 0.0f;
    v = (v >= 1.0f) ? 0.0f : v;
    spk[off] = s;
  }
}

// ---------------------------- transpose-conv 4x4 stride2 pad1 (torch semantics)
// optional fused sigmoid(scale*(x-bias)) when oscale != nullptr
__global__ void deconv4x4_s2_kernel(const float* __restrict__ in, const float* __restrict__ w,
                                    float* __restrict__ out,
                                    int N, int Cin, int Hin, int Win, int Cout,
                                    const float* oscale, const float* obias) {
  const int Ho = Hin * 2, Wo = Win * 2;
  int idx = blockIdx.x * blockDim.x + threadIdx.x;
  int total = N * Cout * Ho * Wo;
  if (idx >= total) return;
  int ox = idx % Wo;
  int oy = (idx / Wo) % Ho;
  int co = (idx / (Wo * Ho)) % Cout;
  int n  = idx / (Wo * Ho * Cout);
  float acc = 0.f;
  for (int ky = 0; ky < 4; ++ky) {
    int ty = oy + 1 - ky;
    if (ty & 1) continue;
    int iy = ty >> 1;
    if (iy < 0 || iy >= Hin) continue;
    for (int kx = 0; kx < 4; ++kx) {
      int tx = ox + 1 - kx;
      if (tx & 1) continue;
      int ix = tx >> 1;
      if (ix < 0 || ix >= Win) continue;
      for (int ci = 0; ci < Cin; ++ci)
        acc += in[((n * Cin + ci) * Hin + iy) * Win + ix] * w[((ci * Cout + co) * 4 + ky) * 4 + kx];
    }
  }
  if (oscale) {
    float zz = (*oscale) * (acc - (*obias));
    out[idx] = 1.0f / (1.0f + expf(-zz));
  } else {
    out[idx] = acc;
  }
}

// LIF scan over deconv1 output + fold linear leaky-integrator weights:
// v_T of leaky integrator = sum_t u_t * 2^(t-16); deconv2 is linear so we pre-sum.
__global__ void lif_wsum_kernel(const float* __restrict__ u, float* __restrict__ wsum) {
  int idx = blockIdx.x * blockDim.x + threadIdx.x;
  const int CHW = 2 * 4096;
  if (idx >= BATCH * CHW) return;
  int cp = idx % CHW;
  int b  = idx / CHW;
  float v = 0.f, acc = 0.f, coef = 0x1p-16f;  // 2^(t-16), t=0
  for (int t = 0; t < T; ++t) {
    float x = u[(t * BATCH + b) * CHW + cp];
    v = 0.5f * (v + x);
    float s = (v >= 1.0f) ? 1.0f : 0.0f;
    v = (v >= 1.0f) ? 0.0f : v;
    acc += s * coef;
    coef *= 2.0f;
  }
  wsum[b * CHW + cp] = acc;   // [32,2,64,64]
}

// ---------------------------------------------------------------- launcher
extern "C" void kernel_launch(void* const* d_in, const int* in_sizes, int n_in,
                              void* d_out, int out_size, void* d_ws, size_t ws_size,
                              hipStream_t stream) {
  const float* x       = (const float*)d_in[0];
  const float* conv1_w = (const float*)d_in[1];
  const float* bn1_g   = (const float*)d_in[2];
  const float* bn1_b   = (const float*)d_in[3];
  const float* conv2_w = (const float*)d_in[4];
  const float* bn2_g   = (const float*)d_in[5];
  const float* bn2_b   = (const float*)d_in[6];
  const float* rc_w    = (const float*)d_in[7];
  const float* rc_b    = (const float*)d_in[8];
  const float* dec_w   = (const float*)d_in[9];
  const float* dct1_w  = (const float*)d_in[10];
  const float* dct2_w  = (const float*)d_in[11];
  const float* o_scale = (const float*)d_in[12];
  const float* o_bias  = (const float*)d_in[13];

  char* base = (char*)d_ws;
  size_t off = 0;
  auto alloc = [&](size_t bytes) { size_t o = off; off += (bytes + 255) & ~(size_t)255; return o; };
  const size_t o_conv1 = alloc(512ull * 2 * 4096 * 4);   // 16 MB (reused: dec_z)
  const size_t o_spk1  = alloc(512ull * 2 * 4096 * 4);   // 16 MB (reused: dec_spk)
  const size_t o_conv2 = alloc(512ull * 4 * 1024 * 4);   // 8 MB  (reused: wsum)
  const size_t o_encsp = alloc(512ull * 4096 * 2);       // 4 MB bf16 spikes
  const size_t o_lat   = alloc(512ull * 4096 * 2);       // 4 MB bf16 spikes
  const size_t o_rcw   = alloc(4096ull * 8192 * 2);      // 64 MB (reused: dc1_out 16 MB)
  const size_t o_decw  = alloc(4096ull * 4096 * 2);      // 32 MB
  const size_t o_stats = alloc(64);                      // 16 floats
  const size_t o_vrc   = alloc(32ull * 4096 * 4);
  const size_t o_y0    = alloc(32ull * 4096 * 2);
  const size_t zero_bytes = off - o_stats;

  float*  conv1_out = (float*)(base + o_conv1);
  float*  spk1      = (float*)(base + o_spk1);
  float*  conv2_out = (float*)(base + o_conv2);
  __bf16* enc_spk   = (__bf16*)(base + o_encsp);
  __bf16* lat       = (__bf16*)(base + o_lat);
  __bf16* rcw_b     = (__bf16*)(base + o_rcw);
  __bf16* decw_b    = (__bf16*)(base + o_decw);
  float*  stats     = (float*)(base + o_stats);
  float*  v_rc      = (float*)(base + o_vrc);
  __bf16* y0        = (__bf16*)(base + o_y0);
  float*  dec_z     = conv1_out;                  // alias (conv1_out dead)
  float*  dec_spk   = spk1;                       // alias (spk1 dead)
  float*  dc1_out   = (float*)(base + o_rcw);     // alias (rc weights dead)
  float*  wsum      = conv2_out;                  // alias (conv2_out dead)

  // zero stats + recurrent membrane state + y[-1]
  {
    int n = (int)(zero_bytes / 4);
    zero_u32_kernel<<<(n + 255) / 256, 256, 0, stream>>>((unsigned int*)(base + o_stats), n);
  }
  // weight conversion to bf16 (L2-resident reuse across 16 recurrent steps)
  cvt_bf16_kernel<<<131072, 256, 0, stream>>>(rc_w,  rcw_b,  4096 * 8192);
  cvt_bf16_kernel<<<65536,  256, 0, stream>>>(dec_w, decw_b, 4096 * 4096);

  // encoder stage 1: conv(1->2,s2) + BN stats + fused BN+LIF scan
  conv3x3_s2_kernel<<<16384, 256, 0, stream>>>(x, conv1_w, conv1_out, 512, 1, 128, 128, 2);
  chan_stats_kernel<<<dim3(256, 2), 256, 0, stream>>>(conv1_out, 512, 2, 4096, stats + 0, stats + 2);
  bn_lif_kernel<<<1024, 256, 0, stream>>>(conv1_out, stats + 0, stats + 2, bn1_g, bn1_b,
                                          2, 4096, 512.0f * 4096.0f, spk1, nullptr);
  // encoder stage 2: conv(2->4,s2) + BN + LIF -> bf16 spikes [T,B,4096]
  conv3x3_s2_kernel<<<8192, 256, 0, stream>>>(spk1, conv2_w, conv2_out, 512, 2, 64, 64, 4);
  chan_stats_kernel<<<dim3(256, 4), 256, 0, stream>>>(conv2_out, 512, 4, 1024, stats + 4, stats + 8);
  bn_lif_kernel<<<512, 256, 0, stream>>>(conv2_out, stats + 4, stats + 8, bn2_g, bn2_b,
                                         4, 1024, 512.0f * 1024.0f, nullptr, enc_spk);

  // recurrent LIF bottleneck: 16 sequential WMMA GEMM + LIF steps
  for (int t = 0; t < T; ++t) {
    const __bf16* yprev = (t == 0) ? y0 : (lat + (size_t)(t - 1) * BATCH * 4096);
    gemm_rc_lif_kernel<<<256, 32, 0, stream>>>(enc_spk + (size_t)t * BATCH * 4096, yprev,
                                               rcw_b, rc_b, v_rc,
                                               lat + (size_t)t * BATCH * 4096);
  }

  // linear decoder (WMMA) + LIF scan
  gemm_dec_kernel<<<dim3(256, 16), 32, 0, stream>>>(lat, decw_b, dec_z);
  lif_dec_kernel<<<512, 256, 0, stream>>>(dec_z, dec_spk);

  // deconv decoder
  deconv4x4_s2_kernel<<<16384, 256, 0, stream>>>(dec_spk, dct1_w, dc1_out,
                                                 512, 4, 32, 32, 2, nullptr, nullptr);
  lif_wsum_kernel<<<1024, 256, 0, stream>>>(dc1_out, wsum);
  // final deconv on the leaky-integrator-weighted sum, fused sigmoid -> d_out [32,1,128,128]
  deconv4x4_s2_kernel<<<2048, 256, 0, stream>>>(wsum, dct2_w, (float*)d_out,
                                                32, 2, 64, 64, 1, o_scale, o_bias);
  (void)in_sizes; (void)n_in; (void)out_size; (void)ws_size;
}
